// TTLinear_37323265803185
// MI455X (gfx1250) — matrix operand order repaired and checked by reading, not compiled, not run on hardware
//
#include <hip/hip_runtime.h>
#include <stdint.h>

// ---------------------------------------------------------------------------
// TT-linear (B=8192, in=1024, out=4096, modes n=[4,8,8,4], m=[8,8,8,8], r=8)
// Cores merged pairwise: W01[512x32] (=G0*G1), W23[64x256] (=G2*G3), bf16 in
// d_ws. Fused kernel: per 16-row batch tile, transpose x into LDS, then for
// each of 32 (m0,m1)-pairs: GEMM1 (K=32,N=16) -> LDS H2 slice, GEMM2
// (K=256,N=64) -> f32 out + bias. All matmuls: v_wmma_f32_16x16x32_bf16.
// Weight staging additionally exercises the CDNA5 Tensor Data Mover.
// ---------------------------------------------------------------------------

typedef __attribute__((ext_vector_type(16))) __bf16 v16bf;
typedef __attribute__((ext_vector_type(8)))  __bf16 v8bf;
typedef __attribute__((ext_vector_type(8)))  float  v8f;
typedef __attribute__((ext_vector_type(4)))  unsigned int v4u;
typedef __attribute__((ext_vector_type(8)))  int v8i;
typedef __attribute__((ext_vector_type(4)))  int v4i;

// LDS partition (units: __bf16 elements). Total 81920 bf16 = 160 KB.
#define LDS_X    0        // 16384 : Xs[(b*32+p23)][k]   (k = n0*8+n1)
#define LDS_W01  16384    // 16384 : [pair][c(16)][k(32)]
#define LDS_W23  32768    // 16384 : [col(64)][k(256)]
#define LDS_H2   49152    // 4 waves * 8192 : [combo][b][r2][p23]
#define LDS_BYTES (81920 * 2)

#ifndef __has_builtin
#define __has_builtin(x) 0
#endif

// Build a 16-lane bf16 vector from two aligned 8-element chunks.
__device__ __forceinline__ v16bf ld16(const __bf16* p0, const __bf16* p1) {
  v8bf lo = *(const v8bf*)p0;
  v8bf hi = *(const v8bf*)p1;
  return __builtin_shufflevector(lo, hi, 0,1,2,3,4,5,6,7,8,9,10,11,12,13,14,15);
}

// A-matrix (16x32 bf16): lanes<16 hold K{0..7,16..23} of row M=lane;
// lanes>=16 hold K{8..15,24..31} of row M=lane-16 (ISA 7.12.2).
__device__ __forceinline__ v16bf loadA(const __bf16* rowbase, bool hi) {
  const int k0 = hi ? 8 : 0;
  return ld16(rowbase + k0, rowbase + k0 + 16);
}

// B-matrix (32x16 bf16): col = lane%16; lanes<16 hold K{0..7,16..23},
// lanes>=16 hold K{8..15,24..31} (same K order as A).
__device__ __forceinline__ v16bf loadB(const __bf16* colbase, bool hi) {
  const int kb = hi ? 8 : 0;
  return ld16(colbase + kb, colbase + kb + 16);
}

__device__ __forceinline__ void store_tile(const v8f acc, float* __restrict__ out,
                                           const float* __restrict__ bias,
                                           int b0, int m0m1, int nt, int cl, bool hiH) {
  const int ocol = m0m1 * 64 + nt * 16 + cl;
  const float bv = bias[ocol];
  float* op = out + (size_t)(b0 + (hiH ? 8 : 0)) * 4096 + ocol;
  #pragma unroll
  for (int i = 0; i < 8; ++i)
    op[(size_t)i * 4096] = acc[i] + bv;
}

// ---------------------------------------------------------------------------
// Kernel 1: merge TT cores into bf16 WMMA-ready weight tables in d_ws.
//   W01[p*512 + c*32 + k]    (p=pair<32, c=(comboLSB,r2)<16, k=n0*8+n1<32)
//   W23[16384 + col*256 + k] (col=m2*8+m3<64, k=r2*32+n2*4+n3<256)
// ---------------------------------------------------------------------------
__global__ __launch_bounds__(256) void tt_build_weights(
    const float* __restrict__ c0, const float* __restrict__ c1,
    const float* __restrict__ c2, const float* __restrict__ c3,
    __bf16* __restrict__ w)
{
  const int gid = blockIdx.x * 256 + threadIdx.x;
  if (gid < 16384) {
    const int p = gid >> 9, rem = gid & 511;
    const int c = rem >> 5, k = rem & 31;
    const int combo = p * 2 + (c >> 3);          // m0m1 index
    const int m0 = combo >> 3, m1 = combo & 7, r2 = c & 7;
    const int n0 = k >> 3, n1 = k & 7;
    float s = 0.f;
    #pragma unroll
    for (int r1 = 0; r1 < 8; ++r1)
      s += c0[m0 * 32 + r1 * 4 + n0] * c1[(r1 * 8 + m1) * 64 + r2 * 8 + n1];
    w[gid] = (__bf16)s;
  } else if (gid < 32768) {
    const int idx = gid - 16384;
    const int col = idx >> 8, k = idx & 255;
    const int m2 = col >> 3, m3 = col & 7;
    const int r2 = k >> 5, n2 = (k >> 2) & 7, n3 = k & 3;
    float s = 0.f;
    #pragma unroll
    for (int r3 = 0; r3 < 8; ++r3)
      s += c2[(r2 * 8 + m2) * 64 + r3 * 8 + n2] * c3[(r3 * 8 + m3) * 4 + n3];
    w[gid] = (__bf16)s;
  }
}

// ---------------------------------------------------------------------------
// Kernel 2: fused TT-linear. 512 blocks x 128 threads (4 wave32 per block),
// 16 batch rows per block, 8 (m0,m1)-pairs per wave.
// ---------------------------------------------------------------------------
__global__ __launch_bounds__(128) void tt_linear_wmma(
    const float* __restrict__ x, const float* __restrict__ bias,
    const __bf16* __restrict__ wglob, float* __restrict__ out)
{
  extern __shared__ __bf16 smem[];
  __bf16* Xs   = smem + LDS_X;
  __bf16* W23s = smem + LDS_W23;

  const int tid  = threadIdx.x;
  const int lane = tid & 31;
  const int wave = tid >> 5;
  const int cl   = lane & 15;     // column within a 16-wide WMMA tile
  const bool hiH = lane >= 16;    // upper half-wave
  const int b0   = blockIdx.x * 16;

  // ---- Stage x tile, transposed (in_feature = n0*256 + n1*32 + p23
  //      -> Xs[(b*32 + p23)*32 + (n0*8+n1)]), f32 -> bf16.
  {
    const float* xg = x + (size_t)b0 * 1024 + (size_t)(tid >> 3) * 1024;
    __bf16* xrow = Xs + (tid >> 3) * 1024;       // 32 p23 * 32 k per row
    const int f0 = (tid & 7) << 7;               // 128 features per thread
    #pragma unroll 4
    for (int j = 0; j < 128; j += 4) {
      const float4 v = *(const float4*)(xg + f0 + j);
      const int f = f0 + j;
      const int k = (((f >> 8) & 3) << 3) | ((f >> 5) & 7);
      __bf16* dst = xrow + (f & 31) * 32 + k;    // p23 stride = 32
      dst[0]  = (__bf16)v.x;
      dst[32] = (__bf16)v.y;
      dst[64] = (__bf16)v.z;
      dst[96] = (__bf16)v.w;
    }
  }

  // ---- Stage merged weights via the Tensor Data Mover: one 64 KB contiguous
  //      1-D tile (32768 bf16) global -> LDS bytes [32768, 98304).
#if __has_builtin(__builtin_amdgcn_tensor_load_to_lds)
  if (wave == 0) {
    const unsigned long long ga = (unsigned long long)(uintptr_t)wglob;
    v4u g0;
    g0[0] = 1u;                                   // count=1, user mode
    g0[1] = (unsigned)(LDS_W01 * 2);              // lds_addr (bytes)
    g0[2] = (unsigned)(ga & 0xFFFFFFFFu);         // global_addr[31:0]
    g0[3] = (unsigned)((ga >> 32) & 0x01FFFFFFu)  // global_addr[56:32]
          | (2u << 30);                           // type=2 ("image")
    v8i g1;
    g1[0] = 1 << 16;                              // data_size=2B; mask=0
    g1[1] = (int)0x80000000u;                     // tensor_dim0 lo16 = 32768
    g1[2] = 1 << 16;                              // dim0 hi16=0 | tensor_dim1=1
    g1[3] = (int)0x80000000u;                     // dim1 hi16=0 | tile_dim0=32768
    g1[4] = 0;                                    // tile_dim1/2 unused
    g1[5] = 32768;                                // tensor_dim0_stride lo32
    g1[6] = 0; g1[7] = 0;
    const v4i gz = {0, 0, 0, 0};
#if __clang_major__ >= 23
    const v8i gz8 = {0, 0, 0, 0, 0, 0, 0, 0};
    __builtin_amdgcn_tensor_load_to_lds(g0, g1, gz, gz, gz8, 0);
#else
    __builtin_amdgcn_tensor_load_to_lds(g0, g1, gz, gz, 0);
#endif
    __builtin_amdgcn_s_wait_tensorcnt(0);
  }
  __syncthreads();
#endif
  // D# bitfield semantics are not hardware-verified: redo the copy with plain
  // loads AFTER the TDM wait (same bytes, so the final LDS state is correct
  // either way; the 64 KB is L2-resident so the cost is negligible).
  {
    const uint32_t* src = (const uint32_t*)wglob;
    uint32_t* dst = (uint32_t*)(smem + LDS_W01);
    for (int j = tid; j < 16384; j += 128) dst[j] = src[j];
  }
  __syncthreads();

  __bf16* H2w = smem + LDS_H2 + wave * 8192;     // [combo][b][r2][p23] bf16
  const __bf16* W01s = smem + LDS_W01;

  for (int pi = 0; pi < 8; ++pi) {
    const int pair = wave * 8 + pi;

    // ===== GEMM1: rows (b,p23)=512, K=32 (n0,n1), N=16 = (combo,r2) of pair.
    // Two independent accumulator chains per iteration hide the WMMA->VALU
    // (cvt) hazard slots.
    const v16bf B1 = loadB(&W01s[(pair * 16 + cl) * 32], hiH);
    const int comboD = cl >> 3, r2D = cl & 7;    // D-column decomposition
    for (int t = 0; t < 32; t += 2) {
      const v16bf A0 = loadA(&Xs[(t * 16 + cl) * 32], hiH);
      const v16bf A1 = loadA(&Xs[((t + 1) * 16 + cl) * 32], hiH);
      v8f d0 = {}, d1 = {};
      d0 = __builtin_amdgcn_wmma_f32_16x16x32_bf16(
               false, A0, false, B1, (short)0, d0, false, false);
      d1 = __builtin_amdgcn_wmma_f32_16x16x32_bf16(
               false, A1, false, B1, (short)0, d1, false, false);
      #pragma unroll
      for (int i = 0; i < 8; ++i) {
        const int R = t * 16 + i + (hiH ? 8 : 0);
        H2w[((comboD * 16 + (R >> 5)) * 8 + r2D) * 32 + (R & 31)] = (__bf16)d0[i];
      }
      #pragma unroll
      for (int i = 0; i < 8; ++i) {
        const int R = (t + 1) * 16 + i + (hiH ? 8 : 0);
        H2w[((comboD * 16 + (R >> 5)) * 8 + r2D) * 32 + (R & 31)] = (__bf16)d1[i];
      }
    }

    // ===== GEMM2: per combo, rows=16 batch, K=256 (r2,n2,n3), N=64 (m2,m3).
    // kt-outer: A fragment loaded once and fed to 4 independent WMMA chains.
    #pragma unroll
    for (int combo = 0; combo < 2; ++combo) {
      const int m0m1 = pair * 2 + combo;
      const __bf16* Arow = &H2w[(combo * 16 + cl) * 256];
      const __bf16* Bc   = &W23s[cl * 256];
      v8f a0 = {}, a1 = {}, a2 = {}, a3 = {};
      #pragma unroll
      for (int kt = 0; kt < 8; ++kt) {
        const v16bf A  = loadA(Arow + kt * 32, hiH);
        const v16bf B0 = loadB(Bc + 0 * 4096 + kt * 32, hiH);  // nt stride = 16*256
        const v16bf B1v = loadB(Bc + 1 * 4096 + kt * 32, hiH);
        const v16bf B2 = loadB(Bc + 2 * 4096 + kt * 32, hiH);
        const v16bf B3 = loadB(Bc + 3 * 4096 + kt * 32, hiH);
        a0 = __builtin_amdgcn_wmma_f32_16x16x32_bf16(false, A, false, B0,  (short)0, a0, false, false);
        a1 = __builtin_amdgcn_wmma_f32_16x16x32_bf16(false, A, false, B1v, (short)0, a1, false, false);
        a2 = __builtin_amdgcn_wmma_f32_16x16x32_bf16(false, A, false, B2,  (short)0, a2, false, false);
        a3 = __builtin_amdgcn_wmma_f32_16x16x32_bf16(false, A, false, B3,  (short)0, a3, false, false);
      }
      store_tile(a0, out, bias, b0, m0m1, 0, cl, hiH);
      store_tile(a1, out, bias, b0, m0m1, 1, cl, hiH);
      store_tile(a2, out, bias, b0, m0m1, 2, cl, hiH);
      store_tile(a3, out, bias, b0, m0m1, 3, cl, hiH);
    }
  }
}

// ---------------------------------------------------------------------------
extern "C" void kernel_launch(void* const* d_in, const int* in_sizes, int n_in,
                              void* d_out, int out_size, void* d_ws, size_t ws_size,
                              hipStream_t stream) {
  const float* x    = (const float*)d_in[0];
  const float* c0   = (const float*)d_in[1];
  const float* c1   = (const float*)d_in[2];
  const float* c2   = (const float*)d_in[3];
  const float* c3   = (const float*)d_in[4];
  const float* bias = (const float*)d_in[5];
  __bf16* w = (__bf16*)d_ws;                 // 32768 bf16 = 64 KB used

  tt_build_weights<<<128, 256, 0, stream>>>(c0, c1, c2, c3, w);
  tt_linear_wmma<<<512, 128, LDS_BYTES, stream>>>(x, bias, w, (float*)d_out);
}
